// Resnet50Encoder_38431367365095
// MI455X (gfx1250) — matrix operand order repaired and checked by reading, not compiled, not run on hardware
//
#include <hip/hip_runtime.h>
#include <hip/hip_bf16.h>

// ---------------------------------------------------------------------------
// Problem constants (from the reference)
// ---------------------------------------------------------------------------
#define TFRAMES 8
#define CIN     1024
#define KWIN    4
#define DDIM    4
#define HDIM    14
#define WDIM    14
#define HW      196
#define NSPAT   784          // 4*14*14
#define KRED    27648        // 27*1024, GEMM reduction (repacked: kk = r*1024 + ci)
#define NIT     864          // KRED / 32
#define COUT1   1024
#define COUT2   512

// GEMM tiling: block = 256 threads = 8 wave32; tile 128(M) x 32(N), K-step 32
#define KSTEP   32
#define MTILE   128
#define NTILE   32
#define NBUF    3            // triple-buffered LDS pipeline (single barrier/iter)

typedef __attribute__((ext_vector_type(16))) __bf16 v16bf;
typedef __attribute__((ext_vector_type(8)))  __bf16 v8bf;
typedef __attribute__((ext_vector_type(8)))  float  v8f;
typedef __attribute__((ext_vector_type(4)))  unsigned short us4;

__device__ __forceinline__ unsigned short f32_to_bf16_bits(float f) {
  unsigned int u = __float_as_uint(f);
  return (unsigned short)((u + 0x7FFFu + ((u >> 16) & 1u)) >> 16);
}

// ---------------------------------------------------------------------------
// Repack conv weight f32 [Cout][Cin][27] -> bf16 [Cout][27][Cin]
// (reduction order r-major/ci-minor so each K-step of 32 has a uniform tap r)
// ---------------------------------------------------------------------------
__global__ void repack_w_bf16(const float* __restrict__ src,
                              unsigned short* __restrict__ dst, int total) {
  int i = blockIdx.x * blockDim.x + threadIdx.x;
  if (i >= total) return;
  int co  = i / KRED;
  int rem = i - co * KRED;
  int r   = rem >> 10;          // 0..26  (tap index kd*9+kh*3+kw)
  int ci  = rem & 1023;
  dst[i] = f32_to_bf16_bits(src[(size_t)co * KRED + (size_t)ci * 27 + r]);
}

// ---------------------------------------------------------------------------
// Build bf16 temporal window: win[t][c][k][hw] = videos[t+k-4][c][hw] or 0
// ---------------------------------------------------------------------------
__global__ void build_win(const float* __restrict__ videos,
                          unsigned short* __restrict__ win, int n) {
  int i = blockIdx.x * blockDim.x + threadIdx.x;
  if (i >= n) return;
  int hw = i % HW;
  int t1 = i / HW;
  int k  = t1 & 3;
  int t2 = t1 >> 2;
  int c  = t2 & (CIN - 1);
  int t  = t2 >> 10;
  int f  = t + k - KWIN;
  float v = (f >= 0) ? videos[((size_t)f * CIN + c) * HW + hw] : 0.0f;
  win[i] = f32_to_bf16_bits(v);
}

// ---------------------------------------------------------------------------
// Implicit-GEMM conv3d (3x3x3 SAME) via BF16 WMMA, f32 accumulate.
// A (MxK) = repacked weights [Cout][27*1024]; B (KxN) = on-the-fly im2col.
// Triple-buffered LDS; A staged with global_load_async_to_lds_b128 (ASYNCcnt),
// B gathered to VGPRs (needs zero-padding) and ds-stored.
// ---------------------------------------------------------------------------
__global__ __launch_bounds__(256) void conv_gemm_wmma(
    const unsigned short* __restrict__ Wb,   // [Cout][KRED] bf16, repacked
    const unsigned short* __restrict__ In,   // [T][CIN][DDIM][HW] bf16
    const float* __restrict__ bias,
    unsigned short* __restrict__ OutBf,      // [T][Cout][NSPAT] (if writeBf16)
    float* __restrict__ OutF,                // [T][Cout][NSPAT] (else)
    int Cout, int writeBf16)
{
  __shared__ unsigned short As[NBUF][MTILE * KSTEP];  // 3 x 8KB
  __shared__ unsigned short Bs[NBUF][NTILE * KSTEP];  // 3 x 2KB

  const int t     = blockIdx.z;
  const int mbase = blockIdx.y * MTILE;
  const int nbase = blockIdx.x * NTILE;     // padded N (800); guard s < 784
  const int tid   = threadIdx.x;
  const int lane  = tid & 31;
  const int wave  = tid >> 5;

  const unsigned short* in_t = In + (size_t)t * CIN * NSPAT;

  // ---- A staging: 2 async b128 per thread per K-step -----------------------
  // chunk0: row m0 = tid>>2 (0..63), 16B at koff; chunk1: row m0+64.
  const unsigned m0      = (unsigned)(tid >> 2);
  const unsigned koffB   = (unsigned)(tid & 3) * 16u;               // bytes
  unsigned aGoff = ((unsigned)(mbase + (int)m0) * (unsigned)KRED) * 2u + koffB;
  const unsigned aGoffD  = 64u * (unsigned)KRED * 2u;               // chunk1 delta
  const unsigned asBase  = (unsigned)(size_t)&As[0][0];             // LDS byte addr
  const unsigned aLds0   = asBase + m0 * 64u + koffB;               // within buf 0
  // per-buffer stride = MTILE*KSTEP*2 = 8192 bytes; chunk1 = +4096 bytes

  // ---- B staging: 4 gathered elems per thread (one column, 4 k) ------------
  const int bq = tid * 4;
  const int bn = bq >> 5;                   // column in tile [0,32)
  const int bk = bq & 31;                   // k0 within tile (multiple of 4)
  const int s  = nbase + bn;
  const int sOK = (s < NSPAT);
  const int d0 = s / HW;
  const int hw0 = s - d0 * HW;
  const int h0 = hw0 / WDIM;
  const int w0 = hw0 - h0 * WDIM;

  // ---- compute-side LDS addressing, contiguous k order (k = 16h + s) ------------
  const int arow = wave * 16 + (lane & 15);
  const int klo  = (lane >> 4) * 16;
  const int bc0  = (lane & 15);
  const int bc1  = bc0 + 16;
  const int bko  = (lane >> 4) * 16;

  v8f acc0 = {};
  v8f acc1 = {};

  // ---------------- staging helpers (as macros to keep asm simple) ----------
#define STAGE_A(bufIdx, gOff)                                                  \
  do {                                                                         \
    unsigned _l0 = aLds0 + (unsigned)(bufIdx) * 8192u;                         \
    asm volatile("global_load_async_to_lds_b128 %0, %1, %2"                    \
                 :: "v"(_l0), "v"(gOff), "s"(Wb) : "memory");                  \
    asm volatile("global_load_async_to_lds_b128 %0, %1, %2"                    \
                 :: "v"(_l0 + 4096u), "v"((gOff) + aGoffD), "s"(Wb)            \
                 : "memory");                                                  \
  } while (0)

#define STAGE_B(bufIdx, itn)                                                   \
  do {                                                                         \
    int _r  = (itn) >> 5;                         /* uniform tap 0..26 */      \
    int _cb = ((itn) & 31) << 5;                  /* ci base */                \
    int _kd = _r / 9;                                                          \
    int _rr = _r - _kd * 9;                                                    \
    int _kh = _rr / 3;                                                         \
    int _kw = _rr - _kh * 3;                                                   \
    int _id = d0 + _kd - 1;                                                    \
    int _ih = h0 + _kh - 1;                                                    \
    int _iw = w0 + _kw - 1;                                                    \
    us4 _v = {0, 0, 0, 0};                                                     \
    if (sOK && (unsigned)_id < DDIM && (unsigned)_ih < HDIM &&                 \
        (unsigned)_iw < WDIM) {                                                \
      size_t _b = (size_t)(_cb + bk) * NSPAT +                                 \
                  (size_t)(_id * HW + _ih * WDIM + _iw);                       \
      _v.x = in_t[_b];                                                         \
      _v.y = in_t[_b + NSPAT];                                                 \
      _v.z = in_t[_b + 2 * NSPAT];                                             \
      _v.w = in_t[_b + 3 * NSPAT];                                             \
    }                                                                          \
    *(us4*)&Bs[bufIdx][bn * KSTEP + bk] = _v;                                  \
  } while (0)

  // ---------------- software pipeline: 1 barrier per iteration --------------
  STAGE_A(0, aGoff);
  STAGE_B(0, 0);
  aGoff += 64u;

  int cbuf = 0, nbuf = 1;
  for (int it = 0; it < NIT; ++it) {
    if (it + 1 < NIT) {
      STAGE_A(nbuf, aGoff);
      STAGE_B(nbuf, it + 1);
      aGoff += 64u;
      // older asyncs (current buffer) complete in order; only the 2 just
      // issued for the next buffer may remain outstanding
      asm volatile("s_wait_asynccnt 0x2" ::: "memory");
    } else {
      asm volatile("s_wait_asynccnt 0x0" ::: "memory");
    }
    __syncthreads();

    // A 16x32: row M = lane&15; lanes<16 K 0..15, lanes>=16 K 16..31
    const unsigned short* Ab = &As[cbuf][0];
    v8bf aLo = *(const v8bf*)&Ab[arow * KSTEP + klo];
    v8bf aHi = *(const v8bf*)&Ab[arow * KSTEP + klo + 8];
    v16bf av = __builtin_shufflevector(aLo, aHi,
        0,1,2,3,4,5,6,7,8,9,10,11,12,13,14,15);

    // B 32x16: col N = lane&15; lanes<16 K 0..15, lanes>=16 K 16..31
    const unsigned short* Bb = &Bs[cbuf][0];
    v8bf b0lo = *(const v8bf*)&Bb[bc0 * KSTEP + bko];
    v8bf b0hi = *(const v8bf*)&Bb[bc0 * KSTEP + bko + 8];
    v16bf bv0 = __builtin_shufflevector(b0lo, b0hi,
        0,1,2,3,4,5,6,7,8,9,10,11,12,13,14,15);
    v8bf b1lo = *(const v8bf*)&Bb[bc1 * KSTEP + bko];
    v8bf b1hi = *(const v8bf*)&Bb[bc1 * KSTEP + bko + 8];
    v16bf bv1 = __builtin_shufflevector(b1lo, b1hi,
        0,1,2,3,4,5,6,7,8,9,10,11,12,13,14,15);

    acc0 = __builtin_amdgcn_wmma_f32_16x16x32_bf16(false, av, false, bv0,
                                                   (short)0, acc0, false, false);
    acc1 = __builtin_amdgcn_wmma_f32_16x16x32_bf16(false, av, false, bv1,
                                                   (short)0, acc1, false, false);

    cbuf = nbuf;
    nbuf = (nbuf == NBUF - 1) ? 0 : nbuf + 1;
  }
#undef STAGE_A
#undef STAGE_B

  // ---- epilogue: bias + ReLU, stores per f32 C-matrix layout ---------------
  const int ncol0 = nbase + (lane & 15);
  const int ncol1 = ncol0 + 16;
#pragma unroll
  for (int v = 0; v < 8; ++v) {
    int co = mbase + wave * 16 + v + 8 * (lane >> 4);
    float b  = bias[co];
    float v0 = acc0[v] + b; v0 = v0 > 0.0f ? v0 : 0.0f;
    float v1 = acc1[v] + b; v1 = v1 > 0.0f ? v1 : 0.0f;
    size_t o0 = ((size_t)t * Cout + co) * NSPAT + ncol0;
    if (writeBf16) {
      OutBf[o0] = f32_to_bf16_bits(v0);
      if (ncol1 < NSPAT) OutBf[o0 + 16] = f32_to_bf16_bits(v1);
    } else {
      OutF[o0] = v0;
      if (ncol1 < NSPAT) OutF[o0 + 16] = v1;
    }
  }
}

// ---------------------------------------------------------------------------
// Global max pool over 784 spatial positions
// ---------------------------------------------------------------------------
__global__ void maxpool_kernel(const float* __restrict__ h2,
                               float* __restrict__ pooled, int total) {
  int i = blockIdx.x * blockDim.x + threadIdx.x;
  if (i >= total) return;
  const float* p = h2 + (size_t)i * NSPAT;
  float m = p[0];
  for (int s2 = 1; s2 < NSPAT; ++s2) m = fmaxf(m, p[s2]);
  pooled[i] = m;
}

// ---------------------------------------------------------------------------
// y[t][o] = relu(sum_c x[t][c] * w[o][c] + b[o])  (tiny, f32)
// ---------------------------------------------------------------------------
__global__ void fc_relu(const float* __restrict__ x, const float* __restrict__ w,
                        const float* __restrict__ b, float* __restrict__ y,
                        int Cin, int Cout) {
  int i = blockIdx.x * blockDim.x + threadIdx.x;
  if (i >= TFRAMES * Cout) return;
  int t = i / Cout;
  int o = i - t * Cout;
  const float* xr = x + (size_t)t * Cin;
  const float* wr = w + (size_t)o * Cin;
  float s = b[o];
  for (int c = 0; c < Cin; ++c) s += xr[c] * wr[c];
  y[i] = s > 0.0f ? s : 0.0f;
}

// ---------------------------------------------------------------------------
// Workspace layout (bytes, 256-aligned); total ~123.6 MB
// ---------------------------------------------------------------------------
#define WS_W1   0u
#define WS_W2   56623104u
#define WS_WIN  84934656u
#define WS_H1   97779712u
#define WS_H2   110624768u
#define WS_POOL 123469824u
#define WS_F1   123486208u
#define WS_F2   123502592u

extern "C" void kernel_launch(void* const* d_in, const int* in_sizes, int n_in,
                              void* d_out, int out_size, void* d_ws, size_t ws_size,
                              hipStream_t stream) {
  const float* videos  = (const float*)d_in[0];
  const float* conv1_w = (const float*)d_in[1];
  const float* conv1_b = (const float*)d_in[2];
  const float* conv2_w = (const float*)d_in[3];
  const float* conv2_b = (const float*)d_in[4];
  const float* l1_w    = (const float*)d_in[5];
  const float* l1_b    = (const float*)d_in[6];
  const float* l2_w    = (const float*)d_in[7];
  const float* l2_b    = (const float*)d_in[8];
  const float* l3_w    = (const float*)d_in[9];
  const float* l3_b    = (const float*)d_in[10];

  char* ws = (char*)d_ws;
  unsigned short* w1b = (unsigned short*)(ws + WS_W1);
  unsigned short* w2b = (unsigned short*)(ws + WS_W2);
  unsigned short* win = (unsigned short*)(ws + WS_WIN);
  unsigned short* h1  = (unsigned short*)(ws + WS_H1);
  float* h2     = (float*)(ws + WS_H2);
  float* pooled = (float*)(ws + WS_POOL);
  float* f1     = (float*)(ws + WS_F1);
  float* f2     = (float*)(ws + WS_F2);

  // 1) repack + cast conv weights to bf16 (tap-major reduction order)
  repack_w_bf16<<<(COUT1 * KRED + 255) / 256, 256, 0, stream>>>(conv1_w, w1b,
                                                               COUT1 * KRED);
  repack_w_bf16<<<(COUT2 * KRED + 255) / 256, 256, 0, stream>>>(conv2_w, w2b,
                                                               COUT2 * KRED);

  // 2) temporal window (bf16), zero-pad + K<->C transpose
  {
    int n = TFRAMES * CIN * KWIN * HW;
    build_win<<<(n + 255) / 256, 256, 0, stream>>>(videos, win, n);
  }

  // 3) conv1: 1024 -> 1024, bf16 out (feeds conv2)
  {
    dim3 grid((NSPAT + NTILE - 1) / NTILE, COUT1 / MTILE, TFRAMES); // (25,8,8)
    conv_gemm_wmma<<<grid, 256, 0, stream>>>(w1b, win, conv1_b, h1, nullptr,
                                             COUT1, 1);
  }

  // 4) conv2: 1024 -> 512, f32 out
  {
    dim3 grid((NSPAT + NTILE - 1) / NTILE, COUT2 / MTILE, TFRAMES); // (25,4,8)
    conv_gemm_wmma<<<grid, 256, 0, stream>>>(w2b, h1, conv2_b, nullptr, h2,
                                             COUT2, 0);
  }

  // 5) global max pool
  maxpool_kernel<<<(TFRAMES * COUT2 + 255) / 256, 256, 0, stream>>>(
      h2, pooled, TFRAMES * COUT2);

  // 6) FC stack
  fc_relu<<<(TFRAMES * 512 + 255) / 256, 256, 0, stream>>>(pooled, l1_w, l1_b, f1, 512, 512);
  fc_relu<<<(TFRAMES * 512 + 255) / 256, 256, 0, stream>>>(f1, l2_w, l2_b, f2, 512, 512);
  fc_relu<<<(TFRAMES * 128 + 255) / 256, 256, 0, stream>>>(f2, l3_w, l3_b, (float*)d_out, 512, 128);
}
